// PENetContext_58222576664668
// MI455X (gfx1250) — compile-verified, matrix operand order checked by reading, not run on hardware
//
#include <hip/hip_runtime.h>
#include <hip/hip_bf16.h>
#include <stdint.h>

// ---------------------------------------------------------------------------
// Problem constants
// ---------------------------------------------------------------------------
#define N_OBJ   2048
#define N_REL   16384
#define C_IN    4096
#define E_DIM   200
#define M_DIM   2048
#define N_CLS   151
#define LN_EPS  1e-5f

typedef __attribute__((ext_vector_type(16))) __bf16 v16bf;
typedef __attribute__((ext_vector_type(2)))  __bf16 v2bf;
typedef __attribute__((ext_vector_type(8)))  float  v8f;

union FragBF {
    v16bf v;
    unsigned int u[8];
};

// pack two f32 -> one dword holding (bf16(lo) | bf16(hi)<<16), RNE rounding.
// Pure 32-bit arithmetic (no 16-bit sub-register ops), merged via v_perm_b32.
__device__ __forceinline__ unsigned int pack_bf16(float lo, float hi) {
#if __has_builtin(__builtin_amdgcn_cvt_pk_bf16_f32)
    v2bf r = __builtin_amdgcn_cvt_pk_bf16_f32(lo, hi);
    return __builtin_bit_cast(unsigned int, r);
#else
    unsigned int ul = __float_as_uint(lo);
    unsigned int uh = __float_as_uint(hi);
    ul += 0x7FFFu + ((ul >> 16) & 1u);
    uh += 0x7FFFu + ((uh >> 16) & 1u);
#if __has_builtin(__builtin_amdgcn_perm)
    // result bytes = { uh[3], uh[2], ul[3], ul[2] }
    return __builtin_amdgcn_perm(uh, ul, 0x07060302u);
#else
    return (ul >> 16) | (uh & 0xFFFF0000u);
#endif
#endif
}

// ---------------------------------------------------------------------------
// Generic bf16-WMMA GEMM: C[M,N] = act(A[M,K] @ W[K,N] + bias (+ C if ADD_C))
// 256 threads = 8 waves, 128x128 C tile, K chunked by 32, double-buffered LDS
// software pipeline (prefetch chunk c+1 in regs across the WMMA burst).
// Wave (wm in 0..3, wn in 0..1) owns a 32x64 sub-tile = 2x4 WMMA tiles.
// ACT: 0 = none, 1 = relu, 2 = sigmoid.  TAIL: K % 32 != 0 (guarded loads).
// Requires: M,N multiples of 128; K multiple of 4; 16B-aligned rows.
// ---------------------------------------------------------------------------
template<int ACT, bool ADD_C, bool TAIL>
__global__ __launch_bounds__(256)
void gemm_bf16_kernel(const float* __restrict__ A, int lda,
                      const float* __restrict__ W, int ldw,
                      const float* __restrict__ bias,
                      float* __restrict__ C, int ldc,
                      int N, int K)
{
    __shared__ unsigned short aT[2][128][34];   // [buf][m][k], skewed stride
    __shared__ unsigned short bT[2][128][34];   // [buf][n][k] (W transposed)

    const int t    = threadIdx.x;
    const int lane = t & 31;
    const int wave = t >> 5;
    const int wm   = wave & 3;   // 4 waves along M (32 rows each)
    const int wn   = wave >> 2;  // 2 waves along N (64 cols each)

    const int bn = blockIdx.x * 128;
    const int bm = blockIdx.y * 128;

    v8f acc[2][4];
#pragma unroll
    for (int mi = 0; mi < 2; ++mi)
#pragma unroll
        for (int ni = 0; ni < 4; ++ni)
#pragma unroll
            for (int r = 0; r < 8; ++r)
                acc[mi][ni][r] = 0.0f;

    const int laneM = lane & 15;
    const int khalf = (lane & 16) ? 8 : 0;

    // loader coordinates
    const int am0 = t >> 2;            // 0..63   (A rows, +64 on 2nd iter)
    const int ak0 = (t & 3) << 3;      // 0,8,16,24 (A k-offset, 8 wide)
    const int wn0 = (t & 31) << 2;     // 0,4,...,124 (W cols, 4 wide)
    const int wkp = (t >> 5) << 1;     // 0,2,...,14  (W k-pair, +16 on 2nd)

    float4 pa[2][2];                   // A prefetch registers
    float4 pw[2][2];                   // W prefetch registers

    auto gload = [&](int kc) {
#pragma unroll
        for (int i = 0; i < 2; ++i) {
            const int m = am0 + (i << 6);
            const float* rowp = A + (size_t)(bm + m) * (size_t)lda + kc + ak0;
            if (!TAIL) {
                pa[i][0] = ((const float4*)rowp)[0];
                pa[i][1] = ((const float4*)rowp)[1];
            } else {
                const float4 z = {0.f, 0.f, 0.f, 0.f};
                pa[i][0] = (kc + ak0     < K) ? ((const float4*)rowp)[0] : z;
                pa[i][1] = (kc + ak0 + 4 < K) ? ((const float4*)rowp)[1] : z;
            }
        }
#pragma unroll
        for (int i = 0; i < 2; ++i) {
            const int kl = wkp + (i << 4);
            const float* p0 = W + (size_t)(kc + kl) * (size_t)ldw + bn + wn0;
            const float* p1 = p0 + ldw;
            if (!TAIL) {
                pw[i][0] = *(const float4*)p0;
                pw[i][1] = *(const float4*)p1;
            } else {
                const float4 z = {0.f, 0.f, 0.f, 0.f};
                pw[i][0] = (kc + kl     < K) ? *(const float4*)p0 : z;
                pw[i][1] = (kc + kl + 1 < K) ? *(const float4*)p1 : z;
            }
        }
    };

    auto stage = [&](int buf) {
#pragma unroll
        for (int i = 0; i < 2; ++i) {
            unsigned int* dst = (unsigned int*)&aT[buf][am0 + (i << 6)][ak0];
            dst[0] = pack_bf16(pa[i][0].x, pa[i][0].y);
            dst[1] = pack_bf16(pa[i][0].z, pa[i][0].w);
            dst[2] = pack_bf16(pa[i][1].x, pa[i][1].y);
            dst[3] = pack_bf16(pa[i][1].z, pa[i][1].w);
        }
#pragma unroll
        for (int i = 0; i < 2; ++i) {
            const int kl = wkp + (i << 4);
            *(unsigned int*)&bT[buf][wn0 + 0][kl] = pack_bf16(pw[i][0].x, pw[i][1].x);
            *(unsigned int*)&bT[buf][wn0 + 1][kl] = pack_bf16(pw[i][0].y, pw[i][1].y);
            *(unsigned int*)&bT[buf][wn0 + 2][kl] = pack_bf16(pw[i][0].z, pw[i][1].z);
            *(unsigned int*)&bT[buf][wn0 + 3][kl] = pack_bf16(pw[i][0].w, pw[i][1].w);
        }
    };

    auto compute = [&](int buf) {
        FragBF afr[2];
        FragBF bfr[4];
#pragma unroll
        for (int mi = 0; mi < 2; ++mi) {
            const unsigned short* base = &aT[buf][wm * 32 + mi * 16 + laneM][0];
#pragma unroll
            for (int v = 0; v < 8; ++v) {
                const int kidx = ((v & 4) ? 16 : 0) + khalf + ((v & 3) << 1);
                afr[mi].u[v] = *(const unsigned int*)(base + kidx);
            }
        }
#pragma unroll
        for (int ni = 0; ni < 4; ++ni) {
            const unsigned short* base = &bT[buf][wn * 64 + ni * 16 + laneM][0];
#pragma unroll
            for (int v = 0; v < 8; ++v) {
                const int kidx = ((v & 4) ? 16 : 0) + khalf + ((v & 3) << 1);
                bfr[ni].u[v] = *(const unsigned int*)(base + kidx);
            }
        }
#pragma unroll
        for (int ni = 0; ni < 4; ++ni)
#pragma unroll
            for (int mi = 0; mi < 2; ++mi)
                acc[mi][ni] = __builtin_amdgcn_wmma_f32_16x16x32_bf16(
                    false, afr[mi].v, false, bfr[ni].v,
                    (short)0, acc[mi][ni], false, false);
    };

    const int nChunks = (K + 31) >> 5;

    // pipeline prologue
    gload(0);
    stage(0);
    __syncthreads();

    int buf = 0;
    for (int c = 0; c < nChunks; ++c) {
        const bool hasNext = (c + 1 < nChunks);
        if (hasNext) gload((c + 1) << 5);   // HBM latency hidden behind WMMAs
        compute(buf);
        if (hasNext) stage(buf ^ 1);
        buf ^= 1;
        __syncthreads();
    }

    // ---- epilogue ----
    const int mAdd = (lane & 16) ? 8 : 0;
#pragma unroll
    for (int mi = 0; mi < 2; ++mi) {
#pragma unroll
        for (int ni = 0; ni < 4; ++ni) {
            const int ng = bn + wn * 64 + ni * 16 + laneM;
            const float bv = bias ? bias[ng] : 0.0f;
#pragma unroll
            for (int r = 0; r < 8; ++r) {
                const int mg = bm + wm * 32 + mi * 16 + r + mAdd;
                const size_t off = (size_t)mg * (size_t)ldc + ng;
                float x = acc[mi][ni][r] + bv;
                if (ADD_C) x += C[off];
                if (ACT == 1) x = fmaxf(x, 0.0f);
                if (ACT == 2) x = 1.0f / (1.0f + __expf(-x));
                C[off] = x;
            }
        }
    }
}

// ---------------------------------------------------------------------------
// entity_dists one-hot + labels-as-float
// ---------------------------------------------------------------------------
__global__ void onehot_labels_kernel(const int* __restrict__ labels,
                                     float* __restrict__ out)
{
    const int idx = blockIdx.x * blockDim.x + threadIdx.x;
    const int oneHotN = N_OBJ * N_CLS;
    if (idx < oneHotN) {
        const int i = idx / N_CLS;
        const int c = idx - i * N_CLS;
        out[idx] = (labels[i] == c) ? 1.0f : 0.0f;
    } else if (idx < oneHotN + N_OBJ) {
        out[idx] = (float)labels[idx - oneHotN];
    }
}

// ---------------------------------------------------------------------------
// ent_embed = obj_embed[labels]
// ---------------------------------------------------------------------------
__global__ void gather_embed_kernel(const float* __restrict__ emb,
                                    const int* __restrict__ labels,
                                    float* __restrict__ ent)
{
    const int i = blockIdx.x;
    const int l = labels[i];
    for (int j = threadIdx.x; j < E_DIM; j += blockDim.x)
        ent[(size_t)i * E_DIM + j] = emb[(size_t)l * E_DIM + j];
}

// ---------------------------------------------------------------------------
// X = X + Y * G   (gated residual, in place)
// ---------------------------------------------------------------------------
__global__ void gate_fma_kernel(float* __restrict__ X,
                                const float* __restrict__ Y,
                                const float* __restrict__ G)
{
    const size_t idx = (size_t)blockIdx.x * blockDim.x + threadIdx.x;
    X[idx] = fmaf(Y[idx], G[idx], X[idx]);
}

// ---------------------------------------------------------------------------
// out = layernorm(T + S0) * g + b    — one block (256 threads) per row of 2048
// ---------------------------------------------------------------------------
__global__ __launch_bounds__(256)
void ln_kernel(const float* __restrict__ T, const float* __restrict__ S0,
               const float* __restrict__ g, const float* __restrict__ b,
               float* __restrict__ out)
{
    const int row = blockIdx.x;
    const int tid = threadIdx.x;
    const size_t base = (size_t)row * M_DIM;

    float x[8];
    float s = 0.0f;
#pragma unroll
    for (int i = 0; i < 8; ++i) {
        const int c = tid + i * 256;
        const float v = T[base + c] + S0[base + c];
        x[i] = v;
        s += v;
    }

    __shared__ float red[256];
    __shared__ float muS, invS;

    red[tid] = s;
    __syncthreads();
    for (int st = 128; st > 0; st >>= 1) {
        if (tid < st) red[tid] += red[tid + st];
        __syncthreads();
    }
    if (tid == 0) muS = red[0] * (1.0f / M_DIM);
    __syncthreads();
    const float mu = muS;

    float ss = 0.0f;
#pragma unroll
    for (int i = 0; i < 8; ++i) {
        const float d = x[i] - mu;
        ss += d * d;
    }
    red[tid] = ss;
    __syncthreads();
    for (int st = 128; st > 0; st >>= 1) {
        if (tid < st) red[tid] += red[tid + st];
        __syncthreads();
    }
    if (tid == 0) invS = rsqrtf(red[0] * (1.0f / M_DIM) + LN_EPS);
    __syncthreads();
    const float inv = invS;

#pragma unroll
    for (int i = 0; i < 8; ++i) {
        const int c = tid + i * 256;
        out[base + c] = (x[i] - mu) * inv * g[c] + b[c];
    }
}

// ---------------------------------------------------------------------------
// fusion_so[r] = relu(SUB[si]+OBJ[oi]) - (SUB[si]-OBJ[oi])^2, one block per rel
// ---------------------------------------------------------------------------
__global__ __launch_bounds__(256)
void fusion_kernel(const float* __restrict__ SUB,
                   const float* __restrict__ OBJ,
                   const int* __restrict__ pair,
                   float* __restrict__ out)
{
    const int r  = blockIdx.x;
    const int si = pair[2 * r + 0];
    const int oi = pair[2 * r + 1];
    const float4* s4 = (const float4*)(SUB + (size_t)si * M_DIM);
    const float4* o4 = (const float4*)(OBJ + (size_t)oi * M_DIM);
    float4* d4 = (float4*)(out + (size_t)r * M_DIM);

#pragma unroll
    for (int it = 0; it < 2; ++it) {
        const int j = threadIdx.x + it * 256;      // 512 float4 per row
        const float4 s = s4[j];
        const float4 o = o4[j];
        float4 res;
        {
            float a = s.x + o.x, d = s.x - o.x; res.x = fmaxf(a, 0.0f) - d * d;
            a = s.y + o.y; d = s.y - o.y;       res.y = fmaxf(a, 0.0f) - d * d;
            a = s.z + o.z; d = s.z - o.z;       res.z = fmaxf(a, 0.0f) - d * d;
            a = s.w + o.w; d = s.w - o.w;       res.w = fmaxf(a, 0.0f) - d * d;
        }
        d4[j] = res;
    }
}

// ---------------------------------------------------------------------------
// Host-side orchestration
// ---------------------------------------------------------------------------
extern "C" void kernel_launch(void* const* d_in, const int* in_sizes, int n_in,
                              void* d_out, int out_size, void* d_ws, size_t ws_size,
                              hipStream_t stream)
{
    (void)in_sizes; (void)n_in; (void)out_size; (void)ws_size;

    const float* roi        = (const float*)d_in[0];
    const int*   labels     = (const int*)  d_in[1];
    const int*   pair       = (const int*)  d_in[2];
    const float* obj_embed  = (const float*)d_in[3];
    const float* post_w     = (const float*)d_in[4];
    const float* post_b     = (const float*)d_in[5];
    const float* wsub1_w    = (const float*)d_in[6];
    const float* wsub1_b    = (const float*)d_in[7];
    const float* wsub2_w    = (const float*)d_in[8];
    const float* wsub2_b    = (const float*)d_in[9];
    const float* wobj1_w    = (const float*)d_in[10];
    const float* wobj1_b    = (const float*)d_in[11];
    const float* wobj2_w    = (const float*)d_in[12];
    const float* wobj2_b    = (const float*)d_in[13];
    const float* gsub_w     = (const float*)d_in[14];
    const float* gsub_b     = (const float*)d_in[15];
    const float* gobj_w     = (const float*)d_in[16];
    const float* gobj_b     = (const float*)d_in[17];
    const float* v2s1_w     = (const float*)d_in[18];
    const float* v2s1_b     = (const float*)d_in[19];
    const float* v2s2_w     = (const float*)d_in[20];
    const float* v2s2_b     = (const float*)d_in[21];
    const float* lsub_w     = (const float*)d_in[22];
    const float* lsub_b     = (const float*)d_in[23];
    const float* lobj_w     = (const float*)d_in[24];
    const float* lobj_b     = (const float*)d_in[25];
    const float* ns_g       = (const float*)d_in[26];
    const float* ns_b       = (const float*)d_in[27];
    const float* no_g       = (const float*)d_in[28];
    const float* no_b       = (const float*)d_in[29];

    float* out = (float*)d_out;
    float* ws  = (float*)d_ws;

    // workspace layout (floats)
    float* REP  = ws;                          // 2048 x 4096
    float* TMP  = REP  + (size_t)N_OBJ * 4096; // 2048 x 4096 (hidden reuse)
    float* ENT  = TMP  + (size_t)N_OBJ * 4096; // 2048 x 200
    float* S    = ENT  + (size_t)N_OBJ * E_DIM;// 2048 x 2048
    float* O    = S    + (size_t)N_OBJ * M_DIM;
    float* SEMS = O    + (size_t)N_OBJ * M_DIM;
    float* SEMO = SEMS + (size_t)N_OBJ * M_DIM;
    float* GS   = SEMO + (size_t)N_OBJ * M_DIM;
    float* GO   = GS   + (size_t)N_OBJ * M_DIM;

    const dim3 blk(256);
    const dim3 gN4096(4096 / 128, N_OBJ / 128);
    const dim3 gN2048(2048 / 128, N_OBJ / 128);
    const dim3 gN1024(1024 / 128, N_OBJ / 128);

    // outputs 0,1: one-hot + labels
    onehot_labels_kernel<<<(N_OBJ * N_CLS + N_OBJ + 255) / 256, blk, 0, stream>>>(labels, out);

    // ent_embed gather
    gather_embed_kernel<<<N_OBJ, blk, 0, stream>>>(obj_embed, labels, ENT);

    // REP = roi @ post_emb + b    (obj_rep = REP[:,0:2048], sub_rep = REP[:,2048:4096])
    gemm_bf16_kernel<0, false, false><<<gN4096, blk, 0, stream>>>(roi, C_IN, post_w, 2 * M_DIM, post_b, REP, 2 * M_DIM, 2 * M_DIM, C_IN);

    // S = mlp2(ENT; wsub)   (per-object, gathered later)
    gemm_bf16_kernel<1, false, true ><<<gN1024, blk, 0, stream>>>(ENT, E_DIM, wsub1_w, M_DIM / 2, wsub1_b, TMP, M_DIM / 2, M_DIM / 2, E_DIM);
    gemm_bf16_kernel<0, false, false><<<gN2048, blk, 0, stream>>>(TMP, M_DIM / 2, wsub2_w, M_DIM, wsub2_b, S, M_DIM, M_DIM, M_DIM / 2);

    // O = mlp2(ENT; wobj)
    gemm_bf16_kernel<1, false, true ><<<gN1024, blk, 0, stream>>>(ENT, E_DIM, wobj1_w, M_DIM / 2, wobj1_b, TMP, M_DIM / 2, M_DIM / 2, E_DIM);
    gemm_bf16_kernel<0, false, false><<<gN2048, blk, 0, stream>>>(TMP, M_DIM / 2, wobj2_w, M_DIM, wobj2_b, O, M_DIM, M_DIM, M_DIM / 2);

    // SEMS = mlp2(sub_rep; v2s)
    gemm_bf16_kernel<1, false, false><<<gN4096, blk, 0, stream>>>(REP + M_DIM, 2 * M_DIM, v2s1_w, 2 * M_DIM, v2s1_b, TMP, 2 * M_DIM, 2 * M_DIM, M_DIM);
    gemm_bf16_kernel<0, false, false><<<gN2048, blk, 0, stream>>>(TMP, 2 * M_DIM, v2s2_w, M_DIM, v2s2_b, SEMS, M_DIM, M_DIM, 2 * M_DIM);

    // SEMO = mlp2(obj_rep; v2s)
    gemm_bf16_kernel<1, false, false><<<gN4096, blk, 0, stream>>>(REP, 2 * M_DIM, v2s1_w, 2 * M_DIM, v2s1_b, TMP, 2 * M_DIM, 2 * M_DIM, M_DIM);
    gemm_bf16_kernel<0, false, false><<<gN2048, blk, 0, stream>>>(TMP, 2 * M_DIM, v2s2_w, M_DIM, v2s2_b, SEMO, M_DIM, M_DIM, 2 * M_DIM);

    // gs = sigmoid([S, SEMS] @ gate_sub + b) via split-K over the concat
    gemm_bf16_kernel<0, false, false><<<gN2048, blk, 0, stream>>>(S, M_DIM, gsub_w, M_DIM, gsub_b, GS, M_DIM, M_DIM, M_DIM);
    gemm_bf16_kernel<2, true , false><<<gN2048, blk, 0, stream>>>(SEMS, M_DIM, gsub_w + (size_t)M_DIM * M_DIM, M_DIM, nullptr, GS, M_DIM, M_DIM, M_DIM);
    // go = sigmoid([O, SEMO] @ gate_obj + b)
    gemm_bf16_kernel<0, false, false><<<gN2048, blk, 0, stream>>>(O, M_DIM, gobj_w, M_DIM, gobj_b, GO, M_DIM, M_DIM, M_DIM);
    gemm_bf16_kernel<2, true , false><<<gN2048, blk, 0, stream>>>(SEMO, M_DIM, gobj_w + (size_t)M_DIM * M_DIM, M_DIM, nullptr, GO, M_DIM, M_DIM, M_DIM);

    // S += SEMS * gs ; O += SEMO * go   (in place)
    gate_fma_kernel<<<(N_OBJ * M_DIM) / 256, blk, 0, stream>>>(S, SEMS, GS);
    gate_fma_kernel<<<(N_OBJ * M_DIM) / 256, blk, 0, stream>>>(O, SEMO, GO);

    // SUB = LN(relu(S @ lin_sub + b) + S)  -> SEMS
    gemm_bf16_kernel<1, false, false><<<gN2048, blk, 0, stream>>>(S, M_DIM, lsub_w, M_DIM, lsub_b, GS, M_DIM, M_DIM, M_DIM);
    ln_kernel<<<N_OBJ, blk, 0, stream>>>(GS, S, ns_g, ns_b, SEMS);
    // OBJ = LN(relu(O @ lin_obj + b) + O)  -> SEMO
    gemm_bf16_kernel<1, false, false><<<gN2048, blk, 0, stream>>>(O, M_DIM, lobj_w, M_DIM, lobj_b, GO, M_DIM, M_DIM, M_DIM);
    ln_kernel<<<N_OBJ, blk, 0, stream>>>(GO, O, no_g, no_b, SEMO);

    // fusion_so (output 2) with the only per-relation gather
    fusion_kernel<<<N_REL, blk, 0, stream>>>(SEMS, SEMO, pair, out + (size_t)N_OBJ * N_CLS + N_OBJ);
}
